// MultiHeadAttention_54915451847062
// MI455X (gfx1250) — matrix operand order repaired and checked by reading, not compiled, not run on hardware
//
#include <hip/hip_runtime.h>

// Problem constants (from reference): B=2, S=2048, HID=1024, NH=16, HD=64
#define Bv   2
#define Sv   2048
#define HIDv 1024
#define NHv  16
#define HDv  64
#define MTOT (Bv * Sv)   // 4096 rows

// GEMM tiling: 128x128 block, 8 waves of 32x64, K-step 32 (bf16 WMMA depth)
#define BM  128
#define BN  128
#define BK  32
#define LDT 40           // LDS tile row stride in ushorts (80B: 16B-aligned rows)

typedef __bf16 bf16_t;
typedef bf16_t v16bf __attribute__((ext_vector_type(16)));
typedef float  v8f   __attribute__((ext_vector_type(8)));
typedef float  v4f   __attribute__((ext_vector_type(4)));
typedef unsigned int v4u __attribute__((ext_vector_type(4)));
typedef unsigned int v2u __attribute__((ext_vector_type(2)));

union FragAB { v16bf v; unsigned int u32[8]; unsigned short u16[16]; };
union FragC  { v8f   v; float f[8]; };
union Pack8  { v4u   v; unsigned short s[8]; };

// ---- CDNA5 async global->LDS path (ASYNCcnt) --------------------------------
#if defined(__has_builtin)
#  if __has_builtin(__builtin_amdgcn_global_load_async_to_lds_b128)
#    define HAVE_ASYNC 1
#  endif
#endif
#ifndef HAVE_ASYNC
#  define HAVE_ASYNC 0
#endif

#if defined(__has_builtin) && __has_builtin(__builtin_amdgcn_s_wait_asynccnt)
#  define WAIT_ASYNC0() __builtin_amdgcn_s_wait_asynccnt(0)
#else
#  define WAIT_ASYNC0() asm volatile("s_wait_asynccnt 0" ::: "memory")
#endif

#if HAVE_ASYNC
// Parameter type per clang diagnostic: 'int __vector(4)' pointers, AS1 (global) / AS3 (LDS)
typedef int v4i_g __attribute__((vector_size(16)));
typedef __attribute__((address_space(1))) v4i_g as1_v4i;
typedef __attribute__((address_space(3))) v4i_g as3_v4i;
__device__ __forceinline__ void async_copy_b128(const void* g, const void* lds) {
  __builtin_amdgcn_global_load_async_to_lds_b128(
      (as1_v4i*)(unsigned long long)(size_t)g,
      (as3_v4i*)(unsigned)(size_t)lds, 0, 0);
}
#endif

// ---- CDNA5 Tensor Data Mover path (TENSORcnt) -------------------------------
#if defined(__has_builtin)
#  if __has_builtin(__builtin_amdgcn_tensor_load_to_lds) && __has_builtin(__builtin_amdgcn_s_wait_tensorcnt)
#    define HAVE_TDM 1
#  endif
#endif
#ifndef HAVE_TDM
#  define HAVE_TDM 0
#endif

#if HAVE_TDM
typedef unsigned int u32x4 __attribute__((vector_size(16)));
typedef int          i32x8 __attribute__((vector_size(32)));
typedef int          i32x4 __attribute__((vector_size(16)));
// 2D tile load: tile_h rows x tile_w elems (2B each), global row stride = row_stride elems,
// LDS rows padded: 32 DWORDs data + 4 DWORDs pad -> 144B stride (kvS[32][72]).
__device__ __forceinline__ void tdm_load_tile_2d(unsigned lds_addr, const void* gaddr,
                                                 int tile_w, int tile_h, int row_stride) {
  unsigned long long ga = (unsigned long long)(size_t)gaddr;
  u32x4 g0;
  g0[0] = 1u;                                        // count=1 (valid user D#)
  g0[1] = lds_addr;                                  // lds_addr[31:0]
  g0[2] = (unsigned)(ga & 0xffffffffu);              // global_addr[95:64]
  g0[3] = (unsigned)((ga >> 32) & 0x01ffffffu)       // global_addr[120:96]
        | (2u << 30);                                // type = 2 ("image")
  i32x8 g1;
  g1[0] = (1 << 16)                                  // data_size = 2 bytes
        | (1 << 20)                                  // pad_enable
        | (4 << 22)                                  // pad_interval: 32 DWORDs
        | (3 << 25);                                 // pad_amount:   4 DWORDs
  g1[1] = (tile_w & 0xffff) << 16;                   // tensor_dim0 lo16
  g1[2] = ((tile_w >> 16) & 0xffff)                  // tensor_dim0 hi16
        | ((tile_h & 0xffff) << 16);                 // tensor_dim1 lo16
  g1[3] = ((tile_h >> 16) & 0xffff)                  // tensor_dim1 hi16
        | ((tile_w & 0xffff) << 16);                 // tile_dim0
  g1[4] = (tile_h & 0xffff);                         // tile_dim1 (tile_dim2 = 0)
  g1[5] = row_stride;                                // tensor_dim0_stride lo32
  g1[6] = 0;                                         // dim0_stride hi16 | dim1_stride lo16
  g1[7] = 0;
  i32x4 z4 = {0, 0, 0, 0};                           // groups 2/3 unused (2D)
  i32x8 z8 = {0, 0, 0, 0, 0, 0, 0, 0};               // 6-arg form: extra group
  __builtin_amdgcn_tensor_load_to_lds(g0, g1, z4, z4, z8, 0);
}
#endif

__device__ __forceinline__ unsigned short f2bf(float x) {
  union { float f; unsigned int u; } c; c.f = x;
  unsigned int u = c.u + 0x7FFFu + ((c.u >> 16) & 1u);  // round-to-nearest-even
  return (unsigned short)(u >> 16);
}

// A-matrix fragment (16xK rows, bf16): lane(0-15)=row, halves hold K 0..15 / 16..31 (contiguous)
__device__ __forceinline__ FragAB load_frag_a(const unsigned short* p, int ld) {
  const int lane = threadIdx.x & 31;
  const unsigned short* rp = p + (lane & 15) * ld + ((lane >> 4) << 4);
  FragAB f;
#pragma unroll
  for (int t = 0; t < 8; ++t) {
    int k = 2 * t;
    f.u32[t] = *(const unsigned int*)(rp + k);
  }
  return f;
}

// B-matrix fragment (Kx16, bf16), source stored column-major (row = output column):
// lane(0-15)=column, lanes 0-15 hold K=0..15, lanes 16-31 hold K=16..31 (contiguous)
__device__ __forceinline__ FragAB load_frag_b(const unsigned short* p, int ld) {
  const int lane = threadIdx.x & 31;
  const unsigned short* rp = p + (lane & 15) * ld + ((lane >> 4) << 4);
  FragAB f;
#pragma unroll
  for (int t = 0; t < 8; ++t) f.u32[t] = *(const unsigned int*)(rp + 2 * t);
  return f;
}

__device__ __forceinline__ void wmma_bf16(FragC& c, const FragAB& a, const FragAB& b) {
  c.v = __builtin_amdgcn_wmma_f32_16x16x32_bf16(false, a.v, false, b.v,
                                                (short)0, c.v, false, false);
}

// ---------------------------------------------------------------------------
// C[M,N] = A[M,K] @ W[K,N] + bias   (bf16 WMMA, f32 accumulate)
// ---------------------------------------------------------------------------
template <bool IN_BF16, bool OUT_BF16>
__global__ __launch_bounds__(256) void gemm_bias(const void* __restrict__ Ain,
                                                 const float* __restrict__ W,
                                                 const float* __restrict__ bias,
                                                 void* __restrict__ Cout,
                                                 int M, int N, int K) {
  __shared__ alignas(16) unsigned short lA [BM][LDT];   // A tile, row-major
  __shared__ alignas(16) unsigned short lBt[BN][LDT];   // W tile, transposed: lBt[n][k]

  const int tid  = threadIdx.x;
  const int lane = tid & 31, wid = tid >> 5;
  const int half = lane >> 4, lcol = lane & 15;
  const int m0 = blockIdx.y * BM, n0 = blockIdx.x * BN;
  const int wm = wid >> 1, wn = wid & 1;    // 4x2 wave grid

  const float*          Af = (const float*)Ain;
  const unsigned short* Ab = (const unsigned short*)Ain;

  FragC acc[2][4];
#pragma unroll
  for (int mi = 0; mi < 2; ++mi)
#pragma unroll
    for (int nj = 0; nj < 4; ++nj)
#pragma unroll
      for (int r = 0; r < 8; ++r) acc[mi][nj].f[r] = 0.f;

  for (int k0 = 0; k0 < K; k0 += BK) {
    // ---- A tile: 128x32 ----
    if constexpr (IN_BF16) {
#pragma unroll
      for (int i = 0; i < 2; ++i) {               // 512 x b128 (verbatim bf16 copy)
        int lin8 = i * 256 + tid;                 // 0..511
        int ar = lin8 >> 2, ak = (lin8 & 3) * 8;
        const void* gsrc = Ab + (size_t)(m0 + ar) * K + k0 + ak;
#if HAVE_ASYNC
        async_copy_b128(gsrc, &lA[ar][ak]);       // global->LDS, no VGPR roundtrip
#else
        *(v4u*)&lA[ar][ak] = *(const v4u*)gsrc;
#endif
      }
    } else {
#pragma unroll
      for (int i = 0; i < 4; ++i) {               // 1024 x float4, convert to bf16
        int lin4 = i * 256 + tid;                 // 0..1023
        int ar = lin4 >> 3, ak = (lin4 & 7) * 4;
        v4f f = *(const v4f*)(Af + (size_t)(m0 + ar) * K + k0 + ak);
        v2u pk;
        pk.x = (unsigned)f2bf(f.x) | ((unsigned)f2bf(f.y) << 16);
        pk.y = (unsigned)f2bf(f.z) | ((unsigned)f2bf(f.w) << 16);
        *(v2u*)&lA[ar][ak] = pk;
      }
    }
    // ---- W tile: 32x128 -> transposed store ----
#pragma unroll
    for (int i = 0; i < 4; ++i) {
      int lin4 = i * 256 + tid;                   // 0..1023
      int bn = (lin4 & 31) * 4, bk = lin4 >> 5;
      v4f g = *(const v4f*)(W + (size_t)(k0 + bk) * N + n0 + bn);
      lBt[bn + 0][bk] = f2bf(g.x);
      lBt[bn + 1][bk] = f2bf(g.y);
      lBt[bn + 2][bk] = f2bf(g.z);
      lBt[bn + 3][bk] = f2bf(g.w);
    }
    if (k0 + BK < K) {                            // prefetch next tiles
      __builtin_prefetch(&W[(size_t)(k0 + BK) * N + n0 + (tid & 127)], 0, 1);
      if constexpr (!IN_BF16)
        __builtin_prefetch(&Af[(size_t)(m0 + (tid >> 1)) * K + k0 + BK], 0, 1);
    }
#if HAVE_ASYNC
    if constexpr (IN_BF16) WAIT_ASYNC0();
#endif
    __syncthreads();

    FragAB a[2], b[4];
#pragma unroll
    for (int mi = 0; mi < 2; ++mi) a[mi] = load_frag_a(&lA[wm * 32 + mi * 16][0], LDT);
#pragma unroll
    for (int nj = 0; nj < 4; ++nj) b[nj] = load_frag_b(&lBt[wn * 64 + nj * 16][0], LDT);
#pragma unroll
    for (int mi = 0; mi < 2; ++mi)
#pragma unroll
      for (int nj = 0; nj < 4; ++nj) wmma_bf16(acc[mi][nj], a[mi], b[nj]);
    __syncthreads();
  }

#pragma unroll
  for (int mi = 0; mi < 2; ++mi) {
#pragma unroll
    for (int nj = 0; nj < 4; ++nj) {
      int gm = m0 + wm * 32 + mi * 16 + 8 * half;  // C layout: row = r + 8*half
      int gn = n0 + wn * 64 + nj * 16 + lcol;
      float bv = bias[gn];
#pragma unroll
      for (int r = 0; r < 8; ++r) {
        float v = acc[mi][nj].f[r] + bv;
        if constexpr (OUT_BF16)
          ((unsigned short*)Cout)[(size_t)(gm + r) * N + gn] = f2bf(v);
        else
          ((float*)Cout)[(size_t)(gm + r) * N + gn] = v;
      }
    }
  }
}

// ---------------------------------------------------------------------------
// Flash attention with q=k=v, scores = floor(qk/8), online softmax.
// Grid: B*NH*(S/128) blocks; 8 waves x 16 query rows each.
// K tile: one TDM tensor_load_to_lds per tile (TENSORcnt), else async b128.
// ---------------------------------------------------------------------------
__global__ __launch_bounds__(256) void flash_attn(const unsigned short* __restrict__ Q,
                                                  unsigned short* __restrict__ Oa) {
  __shared__ alignas(16) unsigned short kvS[32][72];   // K tile row-major (144B rows = TDM pad)
  __shared__ alignas(16) unsigned short vtS[64][34];   // V tile transposed (d, key)
  __shared__ alignas(16) unsigned short pS [8][16][32];// per-wave P scratch

  const int tid  = threadIdx.x;
  const int lane = tid & 31, wid = tid >> 5;
  const int half = lane >> 4, lcol = lane & 15;

  const int qt = blockIdx.x & (Sv / 128 - 1); // 0..15
  const int bh = blockIdx.x >> 4;             // 0..B*NH-1
  const int b  = bh >> 4;                     // NH == 16
  const int h  = bh & (NHv - 1);

  const size_t rowbase = (size_t)b * Sv;
  const unsigned short* Qh = Q + rowbase * HIDv + h * HDv;
  const int q0 = qt * 128 + wid * 16;

  // Persistent Q fragments: rows q0..q0+15, head-dim 0..31 and 32..63
  FragAB qa0 = load_frag_a(Qh + (size_t)q0 * HIDv,      HIDv);
  FragAB qa1 = load_frag_a(Qh + (size_t)q0 * HIDv + 32, HIDv);

  float m_r[8], l_r[8];
  FragC o[4];
#pragma unroll
  for (int r = 0; r < 8; ++r) { m_r[r] = -1e30f; l_r[r] = 0.f; }
#pragma unroll
  for (int nj = 0; nj < 4; ++nj)
#pragma unroll
    for (int r = 0; r < 8; ++r) o[nj].f[r] = 0.f;

  // Per-thread slice of the 32x64 kv tile: one b128 (8 bf16) each
  const int kr = tid >> 3;          // 0..31 (key row)
  const int d8 = (tid & 7) * 8;     // 0,8,...,56 (head-dim chunk)

  for (int kv0 = 0; kv0 < Sv; kv0 += 32) {
    const unsigned short* gsrc = Qh + (size_t)(kv0 + kr) * HIDv + d8;
#if HAVE_TDM
    if (wid == 0) {                              // one TDM DMA for the whole 32x64 tile
      tdm_load_tile_2d((unsigned)(size_t)&kvS[0][0],
                       Qh + (size_t)kv0 * HIDv, HDv, 32, HIDv);
      __builtin_amdgcn_s_wait_tensorcnt(0);
    }
    Pack8 t; t.v = *(const v4u*)gsrc;            // b128 read feeds V^T scatter
#elif HAVE_ASYNC
    async_copy_b128(gsrc, &kvS[kr][d8]);         // global->LDS, no VGPR roundtrip
    Pack8 t; t.v = *(const v4u*)gsrc;
#else
    Pack8 t; t.v = *(const v4u*)gsrc;
    *(v4u*)&kvS[kr][d8] = t.v;
#endif
#pragma unroll
    for (int j = 0; j < 8; ++j) vtS[d8 + j][kr] = t.s[j];
#if !HAVE_TDM && HAVE_ASYNC
    WAIT_ASYNC0();
#endif
    __syncthreads();

    // scores for 32 keys = two 16x16 tiles, each 2 WMMAs over d=64
    FragC sc[2];
#pragma unroll
    for (int sub = 0; sub < 2; ++sub) {
#pragma unroll
      for (int r = 0; r < 8; ++r) sc[sub].f[r] = 0.f;
      FragAB kb0 = load_frag_b(&kvS[sub * 16][0],  72);
      FragAB kb1 = load_frag_b(&kvS[sub * 16][32], 72);
      wmma_bf16(sc[sub], qa0, kb0);
      wmma_bf16(sc[sub], qa1, kb1);
#pragma unroll
      for (int r = 0; r < 8; ++r) sc[sub].f[r] = floorf(sc[sub].f[r] * 0.125f);
    }

    float rm[8], corr[8], rs[8];
#pragma unroll
    for (int r = 0; r < 8; ++r) rm[r] = fmaxf(sc[0].f[r], sc[1].f[r]);
#pragma unroll
    for (int off = 1; off < 16; off <<= 1)    // reduce within 16-lane halves
#pragma unroll
      for (int r = 0; r < 8; ++r) rm[r] = fmaxf(rm[r], __shfl_xor(rm[r], off, 32));
#pragma unroll
    for (int r = 0; r < 8; ++r) {
      float mn = fmaxf(m_r[r], rm[r]);
      corr[r] = __expf(m_r[r] - mn);
      m_r[r] = mn;
      rs[r] = 0.f;
    }
#pragma unroll
    for (int sub = 0; sub < 2; ++sub)
#pragma unroll
      for (int r = 0; r < 8; ++r) {
        float p = __expf(sc[sub].f[r] - m_r[r]);
        sc[sub].f[r] = p;
        rs[r] += p;
      }
#pragma unroll
    for (int off = 1; off < 16; off <<= 1)
#pragma unroll
      for (int r = 0; r < 8; ++r) rs[r] += __shfl_xor(rs[r], off, 32);
#pragma unroll
    for (int r = 0; r < 8; ++r) l_r[r] = l_r[r] * corr[r] + rs[r];

    // C-layout P -> row-major LDS scratch -> A-fragment (in-wave, DS in-order)
#pragma unroll
    for (int sub = 0; sub < 2; ++sub)
#pragma unroll
      for (int r = 0; r < 8; ++r)
        pS[wid][r + 8 * half][sub * 16 + lcol] = f2bf(sc[sub].f[r]);
    asm volatile("s_wait_dscnt 0" ::: "memory");
    FragAB pa = load_frag_a(&pS[wid][0][0], 32);

#pragma unroll
    for (int nj = 0; nj < 4; ++nj)
#pragma unroll
      for (int r = 0; r < 8; ++r) o[nj].f[r] *= corr[r];

#pragma unroll
    for (int nj = 0; nj < 4; ++nj) {          // P(16x32) @ V(32x64)
      FragAB vb = load_frag_b(&vtS[nj * 16][0], 34);
      wmma_bf16(o[nj], pa, vb);
    }
    __syncthreads();
  }

  float inv[8];
#pragma unroll
  for (int r = 0; r < 8; ++r) inv[r] = 1.0f / l_r[r];
#pragma unroll
  for (int nj = 0; nj < 4; ++nj) {
    int col = h * HDv + nj * 16 + lcol;
#pragma unroll
    for (int r = 0; r < 8; ++r) {
      size_t gr = rowbase + q0 + r + 8 * half;
      Oa[gr * HIDv + col] = f2bf(o[nj].f[r] * inv[r]);
    }
  }
}

// ---------------------------------------------------------------------------
extern "C" void kernel_launch(void* const* d_in, const int* in_sizes, int n_in,
                              void* d_out, int out_size, void* d_ws, size_t ws_size,
                              hipStream_t stream) {
  (void)in_sizes; (void)n_in; (void)out_size; (void)ws_size;

  const float* x  = (const float*)d_in[0];
  const float* Wq = (const float*)d_in[1];
  const float* bq = (const float*)d_in[2];
  const float* Wo = (const float*)d_in[3];
  const float* bo = (const float*)d_in[4];

  unsigned short* Qbf = (unsigned short*)d_ws;                 // [4096][1024] bf16 (8 MB)
  unsigned short* Abf = Qbf + (size_t)MTOT * HIDv;             // [4096][1024] bf16 (8 MB)

  dim3 blk(256, 1, 1);
  dim3 g1(HIDv / BN, MTOT / BM, 1);                            // (8, 32)

  // 1) Q = x @ Wq + bq  -> bf16
  gemm_bias<false, true><<<g1, blk, 0, stream>>>(x, Wq, bq, Qbf, MTOT, HIDv, HIDv);

  // 2) flash attention (q=k=v=Q), floor(qk/8) scores -> bf16 attn-out
  flash_attn<<<dim3(Bv * NHv * (Sv / 128), 1, 1), blk, 0, stream>>>(Qbf, Abf);

  // 3) out = attn @ Wo + bo  -> f32
  gemm_bias<true, false><<<g1, blk, 0, stream>>>(Abf, Wo, bo, d_out, MTOT, HIDv, HIDv);
}
